// CAEncoder_8057358647468
// MI455X (gfx1250) — matrix operand order repaired and checked by reading, not compiled, not run on hardware
//
#include <hip/hip_runtime.h>
#include <hip/hip_bf16.h>

// CDNA5 / gfx1250: wave32, WMMA 16x16x32 bf16 -> f32
typedef __attribute__((ext_vector_type(16))) __bf16 v16bf;
typedef __attribute__((ext_vector_type(8)))  float  v8f;

#define BB 32
#define NV 512
#define NT 64
#define DD 512
#define SCALE 0.04419417382415922f  // 1/sqrt(512)

union ABpack {
  v16bf v;
  uint4 q[2];
};

// Convert 16 f32 (A-operand pattern: 8 at ap[0..7], 8 at ap[16..23]) to v16bf.
__device__ __forceinline__ void cvt16_f32_to_bf16(const float* __restrict__ ap, v16bf& dst) {
  float4 f0 = *(const float4*)(ap);
  float4 f1 = *(const float4*)(ap + 4);
  float4 f2 = *(const float4*)(ap + 16);
  float4 f3 = *(const float4*)(ap + 20);
  dst[0]  = (__bf16)f0.x; dst[1]  = (__bf16)f0.y; dst[2]  = (__bf16)f0.z; dst[3]  = (__bf16)f0.w;
  dst[4]  = (__bf16)f1.x; dst[5]  = (__bf16)f1.y; dst[6]  = (__bf16)f1.z; dst[7]  = (__bf16)f1.w;
  dst[8]  = (__bf16)f2.x; dst[9]  = (__bf16)f2.y; dst[10] = (__bf16)f2.z; dst[11] = (__bf16)f2.w;
  dst[12] = (__bf16)f3.x; dst[13] = (__bf16)f3.y; dst[14] = (__bf16)f3.z; dst[15] = (__bf16)f3.w;
}

// ---------------------------------------------------------------------------
// Kernel A: per (v,t) pair compute S = scale * V T^T (512x64), dual softmax,
// write vis_att_scores (m,n) and text_att_scores (n,m).
// ---------------------------------------------------------------------------
__global__ __launch_bounds__(512)
void ca_scores_kernel(const float* __restrict__ video,
                      const float* __restrict__ text,
                      const unsigned char* __restrict__ mask,
                      float* __restrict__ out_vis,     // (v,t,m,n)
                      float* __restrict__ out_text) {  // (v,t,n,m)
  __shared__ __bf16 Tsh[NT][520];     // text, bf16, padded pitch   (66.6 KB)
  __shared__ __bf16 Vsh[64][520];     // video chunk, bf16          (66.6 KB)
  __shared__ float  Ssh[NV][68];      // scaled scores S[m][n]      (139 KB)
  __shared__ float  red[NT][8];
  __shared__ float  mx_n[NT], den_n[NT];
  __shared__ float  mx_m[NV], den_m[NV];
  __shared__ unsigned char msk[NT];

  const int pair = blockIdx.x;
  const int v = pair >> 5, t = pair & 31;
  const int tid = threadIdx.x;
  const float* Vg = video + (size_t)v * NV * DD;
  const float* Tg = text  + (size_t)t * NT * DD;

  // stage T as bf16 (coalesced global reads)
  for (int i = tid; i < NT * DD; i += 512)
    Tsh[i >> 9][i & 511] = (__bf16)Tg[i];
  if (tid < NT) msk[tid] = mask[t * NT + tid];

  const int wave = tid >> 5, lane = tid & 31;
  const int row = lane & 15, hi = lane >> 4;
  const int mt = wave >> 2, nt = wave & 3;   // one 16x16 tile per wave per chunk

  for (int mc = 0; mc < 8; ++mc) {           // 64 video rows per chunk
    __syncthreads();
    for (int i = tid; i < 64 * DD; i += 512)
      Vsh[i >> 9][i & 511] = (__bf16)Vg[(size_t)(mc * 64 + (i >> 9)) * DD + (i & 511)];
    __syncthreads();

    v8f acc = {0.f, 0.f, 0.f, 0.f, 0.f, 0.f, 0.f, 0.f};
#pragma unroll
    for (int k0 = 0; k0 < DD; k0 += 32) {
      ABpack a, b;
      // A: 16x32 bf16; lane-low K=[k0..k0+7, k0+16..k0+23], lane-high +8
      const __bf16* ap = &Vsh[mt * 16 + row][k0 + hi * 8];
      a.q[0] = *(const uint4*)(ap);
      a.q[1] = *(const uint4*)(ap + 16);
      // B: 32x16 bf16; lane-low K=[k0..k0+7, k0+16..k0+23], lane-high +8
      const __bf16* bp = &Tsh[nt * 16 + row][k0 + hi * 8];
      b.q[0] = *(const uint4*)(bp);
      b.q[1] = *(const uint4*)(bp + 16);
      acc = __builtin_amdgcn_wmma_f32_16x16x32_bf16(false, a.v, false, b.v,
                                                    (short)0, acc, false, false);
    }
    const int mbase = mc * 64 + mt * 16 + hi * 8;
#pragma unroll
    for (int i = 0; i < 8; ++i)
      Ssh[mbase + i][nt * 16 + row] = acc[i] * SCALE;
  }
  __syncthreads();

  // --- text-att softmax: over m (512) for each n; 8 partials per n ---
  {
    const int n = tid >> 3, p = tid & 7;
    float mx = -1e30f;
    for (int m = p; m < NV; m += 8) mx = fmaxf(mx, Ssh[m][n]);
    red[n][p] = mx;
    __syncthreads();
    float m8 = -1e30f;
#pragma unroll
    for (int j = 0; j < 8; ++j) m8 = fmaxf(m8, red[n][j]);
    float s = 0.f;
    for (int m = p; m < NV; m += 8) s += __expf(Ssh[m][n] - m8);
    __syncthreads();
    red[n][p] = s;
    __syncthreads();
    float den = 0.f;
#pragma unroll
    for (int j = 0; j < 8; ++j) den += red[n][j];
    if (p == 0) { mx_n[n] = m8; den_n[n] = 1.0f / den; }
  }
  // --- vis-att softmax stats: masked over n (64) for each m; one thread/m ---
  {
    const int m = tid;
    float mx = -1e30f;
    for (int n = 0; n < NT; ++n)
      if (msk[n]) mx = fmaxf(mx, Ssh[m][n]);
    float s = 0.f;
    for (int n = 0; n < NT; ++n)
      if (msk[n]) s += __expf(Ssh[m][n] - mx);
    mx_m[m] = mx;
    den_m[m] = 1.0f / s;
  }
  __syncthreads();

  // write text_att_scores (n,m) coalesced
  float* ot = out_text + (size_t)pair * NT * NV;
  for (int i = tid; i < NT * NV; i += 512) {
    const int n = i >> 9, m = i & 511;
    ot[i] = __expf(Ssh[m][n] - mx_n[n]) * den_n[n];
  }
  // write vis_att_scores (m,n) coalesced
  float* ov = out_vis + (size_t)pair * NV * NT;
  for (int i = tid; i < NV * NT; i += 512) {
    const int m = i >> 6, n = i & 63;
    ov[i] = msk[n] ? __expf(Ssh[m][n] - mx_m[m]) * den_m[m] : 0.f;
  }
}

// ---------------------------------------------------------------------------
// Kernel B1: video_embd_att[v,t,m,d] = sum_n vis_att[v,t,m,n] * T[t,n,d]
// (512x512 output per pair, K=64). A from global (L2-hot), B = T in LDS [d][n].
// ---------------------------------------------------------------------------
__global__ __launch_bounds__(512)
void ca_video_att_kernel(const float* __restrict__ text,
                         const float* __restrict__ vis_att,
                         float* __restrict__ out0) {
  __shared__ __bf16 Tt[DD][72];   // transposed text [d][n], 16B-aligned rows
  const int pair = blockIdx.x;
  const int t = pair & 31;
  const int tid = threadIdx.x;
  const float* Tg = text + (size_t)t * NT * DD;
  for (int i = tid; i < NT * DD; i += 512)
    Tt[i & 511][i >> 9] = (__bf16)Tg[i];
  __syncthreads();

  const int wave = tid >> 5, lane = tid & 31;
  const int row = lane & 15, hi = lane >> 4;
  const float* P2 = vis_att + (size_t)pair * NV * NT;
  float* O = out0 + (size_t)pair * NV * DD;

  for (int ms = 0; ms < 2; ++ms) {              // 2 m-tiles per wave
    const int m0 = (wave * 2 + ms) * 16;
    ABpack a[2];                                 // A reused across all d-tiles
#pragma unroll
    for (int kk = 0; kk < 2; ++kk)
      cvt16_f32_to_bf16(P2 + (size_t)(m0 + row) * NT + kk * 32 + hi * 8, a[kk].v);

    for (int dt = 0; dt < 32; ++dt) {
      const int d0 = dt * 16;
      v8f acc = {0.f, 0.f, 0.f, 0.f, 0.f, 0.f, 0.f, 0.f};
#pragma unroll
      for (int kk = 0; kk < 2; ++kk) {
        ABpack b;
        const __bf16* bp = &Tt[d0 + row][kk * 32 + hi * 8];
        b.q[0] = *(const uint4*)(bp);
        b.q[1] = *(const uint4*)(bp + 16);
        acc = __builtin_amdgcn_wmma_f32_16x16x32_bf16(false, a[kk].v, false, b.v,
                                                      (short)0, acc, false, false);
      }
#pragma unroll
      for (int i = 0; i < 8; ++i)
        O[(size_t)(m0 + i + hi * 8) * DD + d0 + row] = acc[i];
    }
  }
}

// ---------------------------------------------------------------------------
// Kernel B2: text_embd_att[v,t,n,d] = sum_m text_att[v,t,n,m] * V[v,m,d]
// (64x512 output per pair, K=512 in 8 chunks). Accumulators persist in VGPRs.
// ---------------------------------------------------------------------------
__global__ __launch_bounds__(512)
void ca_text_att_kernel(const float* __restrict__ video,
                        const float* __restrict__ text_att,
                        float* __restrict__ out1) {
  __shared__ __bf16 Vt[DD][72];   // transposed video chunk [d][m_local]
  const int pair = blockIdx.x;
  const int v = pair >> 5;
  const int tid = threadIdx.x;
  const int wave = tid >> 5, lane = tid & 31;
  const int row = lane & 15, hi = lane >> 4;
  const float* Vg = video + (size_t)v * NV * DD;
  const float* P1 = text_att + (size_t)pair * NT * NV;
  float* O = out1 + (size_t)pair * NT * DD;

  const int n0 = (wave & 3) * 16;
  const int dgrp = wave >> 2;              // wave owns d-tiles dgrp*8 .. dgrp*8+7

  v8f acc[8];
  const v8f zero = {0.f, 0.f, 0.f, 0.f, 0.f, 0.f, 0.f, 0.f};
#pragma unroll
  for (int j = 0; j < 8; ++j) acc[j] = zero;

  for (int kc = 0; kc < 8; ++kc) {         // K chunks of 64 video rows
    __syncthreads();
    for (int i = tid; i < 64 * DD; i += 512)
      Vt[i & 511][i >> 9] = (__bf16)Vg[(size_t)(kc * 64 + (i >> 9)) * DD + (i & 511)];
    __syncthreads();
#pragma unroll
    for (int kk = 0; kk < 2; ++kk) {
      ABpack a;
      cvt16_f32_to_bf16(P1 + (size_t)(n0 + row) * NV + kc * 64 + kk * 32 + hi * 8, a.v);
#pragma unroll
      for (int j = 0; j < 8; ++j) {
        ABpack b;
        const __bf16* bp = &Vt[(dgrp * 8 + j) * 16 + row][kk * 32 + hi * 8];
        b.q[0] = *(const uint4*)(bp);
        b.q[1] = *(const uint4*)(bp + 16);
        acc[j] = __builtin_amdgcn_wmma_f32_16x16x32_bf16(false, a.v, false, b.v,
                                                         (short)0, acc[j], false, false);
      }
    }
  }
#pragma unroll
  for (int j = 0; j < 8; ++j) {
    const int d0 = (dgrp * 8 + j) * 16;
#pragma unroll
    for (int i = 0; i < 8; ++i)
      O[(size_t)(n0 + i + hi * 8) * DD + d0 + row] = acc[j][i];
  }
}

// ---------------------------------------------------------------------------
extern "C" void kernel_launch(void* const* d_in, const int* in_sizes, int n_in,
                              void* d_out, int out_size, void* d_ws, size_t ws_size,
                              hipStream_t stream) {
  (void)in_sizes; (void)n_in; (void)out_size; (void)d_ws; (void)ws_size;
  const float* video = (const float*)d_in[0];
  const float* text  = (const float*)d_in[1];
  const unsigned char* mask = (const unsigned char*)d_in[2];  // numpy bool
  float* out = (float*)d_out;

  const size_t SZ0 = (size_t)BB * BB * NV * DD;   // video_embd_att
  const size_t SZ1 = (size_t)BB * BB * NT * DD;   // text_embd_att
  const size_t SZ2 = (size_t)BB * BB * NV * NT;   // vis_att_scores
  float* out0 = out;                   // video_embd_att
  float* out1 = out + SZ0;             // text_embd_att
  float* out2 = out + SZ0 + SZ1;       // vis_att_scores
  float* out3 = out + SZ0 + SZ1 + SZ2; // text_att_scores

  dim3 grid(BB * BB), block(512);
  ca_scores_kernel<<<grid, block, 0, stream>>>(video, text, mask, out2, out3);
  ca_video_att_kernel<<<grid, block, 0, stream>>>(text, out2, out0);
  ca_text_att_kernel<<<grid, block, 0, stream>>>(video, out3, out1);
}
